// SelfAttention_42348377538660
// MI455X (gfx1250) — matrix operand order repaired and checked by reading, not compiled, not run on hardware
//
#include <hip/hip_runtime.h>

typedef __attribute__((ext_vector_type(16))) _Float16 v16h;
typedef __attribute__((ext_vector_type(8)))  float    v8f;

union FragH { v16h v; unsigned u[8]; };

// ---------------------------------------------------------------------------
// Workspace layout (bytes), all 256-aligned:
//   Wall  : f16 [320][256]                 163840
//   ball  : f32 [320]                        1280 (+pad)
//   xHT   : f16 [4][4096][256]            8388608
//   fT    : f16 [4][4096][32]             1048576
//   gT    : f16 [4][4096][32]             1048576
//   hC    : f16 [4][256][4096]            8388608
// ---------------------------------------------------------------------------
#define OFF_WALL  0
#define OFF_BALL  (OFF_WALL + 163840)
#define OFF_XHT   (OFF_BALL + 2048)
#define OFF_FT    (OFF_XHT + 8388608)
#define OFF_GT    (OFF_FT + 1048576)
#define OFF_HC    (OFF_GT + 1048576)

// ---------------------------------------------------------------------------
// Kernel 1: pack wf/wg/wh -> Wall f16 [320][256], biases -> ball f32 [320]
// ---------------------------------------------------------------------------
__global__ void pack_w_kernel(const float* __restrict__ wf, const float* __restrict__ bf,
                              const float* __restrict__ wg, const float* __restrict__ bg,
                              const float* __restrict__ wh, const float* __restrict__ bh,
                              _Float16* __restrict__ Wall, float* __restrict__ ball) {
    int r = blockIdx.x;      // 0..319
    int c = threadIdx.x;     // 0..255
    float w, bias;
    if (r < 32)        { w = wf[r * 256 + c];        bias = bf[r];      }
    else if (r < 64)   { w = wg[(r - 32) * 256 + c]; bias = bg[r - 32]; }
    else               { w = wh[(r - 64) * 256 + c]; bias = bh[r - 64]; }
    Wall[r * 256 + c] = (_Float16)w;
    if (c == 0) ball[r] = bias;
}

// ---------------------------------------------------------------------------
// Kernel 2: x f32 [B][256][4096] -> xHT f16 [B][4096][256] (LDS tiled)
// ---------------------------------------------------------------------------
__global__ __launch_bounds__(256) void xpose_kernel(const float* __restrict__ x,
                                                    _Float16* __restrict__ xHT) {
    __shared__ _Float16 tile[32][33];
    int bid = blockIdx.x;          // 4096 blocks: b(4) x ct(8) x nt(128)
    int b   = bid >> 10;
    int rem = bid & 1023;
    int ct  = rem >> 7;
    int nt  = rem & 127;
    int c0 = ct * 32, n0 = nt * 32;
    int tx = threadIdx.x & 31, ty = threadIdx.x >> 5;   // 32 x 8
    const float* xb = x + (size_t)b * 256 * 4096;
#pragma unroll
    for (int q = 0; q < 4; ++q) {
        int c = ty + 8 * q;
        tile[c][tx] = (_Float16)xb[(size_t)(c0 + c) * 4096 + n0 + tx];
    }
    __syncthreads();
    _Float16* ob = xHT + (size_t)b * 4096 * 256;
#pragma unroll
    for (int q = 0; q < 4; ++q) {
        int n = ty + 8 * q;
        ob[(size_t)(n0 + n) * 256 + c0 + tx] = tile[tx][n];
    }
}

// ---------------------------------------------------------------------------
// Kernel 3: projections. P[320x4096] = Wall[320x256] * X[256x4096] + ball.
// One wave per 16x16 output tile; K=256 in 8 WMMA steps.
// Rows 0..31 -> fT [n][32], rows 32..63 -> gT [n][32], rows 64..319 -> hC [c][4096]
// ---------------------------------------------------------------------------
__global__ __launch_bounds__(256) void proj_kernel(const _Float16* __restrict__ Wall,
                                                   const float* __restrict__ ball,
                                                   const _Float16* __restrict__ xHT,
                                                   _Float16* __restrict__ fT,
                                                   _Float16* __restrict__ gT,
                                                   _Float16* __restrict__ hC) {
    const int lane = threadIdx.x & 31;
    const int wave = threadIdx.x >> 5;
    const int gid  = blockIdx.x * 8 + wave;     // 20480 waves = 4 * 20 * 256
    const int b    = gid / (20 * 256);
    const int rem  = gid % (20 * 256);
    const int r0   = (rem / 256) * 16;
    const int n0   = (rem % 256) * 16;
    const int half = lane >> 4, l16 = lane & 15;
    const _Float16* xb = xHT + (size_t)b * 4096 * 256;

    v8f acc = {};
    for (int k0 = 0; k0 < 256; k0 += 32) {
        FragH a, bb;
#pragma unroll
        for (int j = 0; j < 8; ++j) {
            // A (16x32 f16): lane half L<16: K = 0..7,16..23 ; L>=16: +8
            int ka = k0 + 2 * j + (j >= 4 ? 8 : 0) + (half ? 8 : 0);
            a.u[j] = *(const unsigned*)(Wall + (size_t)(r0 + l16) * 256 + ka);
            // B (32x16 f16): lane half L<16: K = 0..7,16..23 ; L>=16: +8
            int kb = k0 + 2 * j + (j >= 4 ? 8 : 0) + (half ? 8 : 0);
            bb.u[j] = *(const unsigned*)(xb + (size_t)(n0 + l16) * 256 + kb);
        }
        acc = __builtin_amdgcn_wmma_f32_16x16x32_f16(false, a.v, false, bb.v,
                                                     (short)0, acc, false, false);
    }
    // C/D layout: lane holds (M = v + 8*half, N = l16)
#pragma unroll
    for (int v = 0; v < 8; ++v) {
        int M = r0 + v + 8 * half;     // global out-row 0..319
        int n = n0 + l16;
        _Float16 hv = (_Float16)(acc[v] + ball[M]);
        if (M < 32)       fT[(size_t)b * 4096 * 32 + (size_t)n * 32 + M] = hv;
        else if (M < 64)  gT[(size_t)b * 4096 * 32 + (size_t)n * 32 + (M - 32)] = hv;
        else              hC[(size_t)b * 256 * 4096 + (size_t)(M - 64) * 4096 + n] = hv;
    }
}

// ---------------------------------------------------------------------------
// Async stage of one 32-column slab into LDS (per 256-thread block):
//   f tile: 32n x 32ch f16 (2 KB)  -> one async b64 per thread
//   h tile: 256c x 32k f16 (16 KB) -> four async b128 per thread
// Tracked by ASYNCcnt; caller pairs with s_wait_asynccnt + barrier.
// ---------------------------------------------------------------------------
__device__ __forceinline__ void stage_async(const _Float16* fsrc, const _Float16* hsrc,
                                            _Float16* fdst, _Float16* hdst, int tid) {
    unsigned           ldsf = (unsigned)(uintptr_t)fdst + tid * 8;
    unsigned long long gf   = (unsigned long long)(uintptr_t)fsrc + tid * 8;
    asm volatile("global_load_async_to_lds_b64 %0, %1, off"
                 :: "v"(ldsf), "v"(gf) : "memory");
    unsigned           ldsh = (unsigned)(uintptr_t)hdst + tid * 64;
    unsigned long long gh   = (unsigned long long)(uintptr_t)(hsrc + (size_t)tid * 4096);
#pragma unroll
    for (int q = 0; q < 4; ++q) {
        asm volatile("global_load_async_to_lds_b128 %0, %1, off"
                     :: "v"(ldsh + q * 16), "v"(gh + q * 16) : "memory");
    }
}

// ---------------------------------------------------------------------------
// Kernel 4: fused flash attention + epilogue.
// 1 wave = 16 query rows. Per 32-col step: 2 s-WMMAs (K=32=INTER), online
// softmax, p through LDS (C/D -> A layout), 16 o-WMMAs over 256 channels.
// h/f tiles double-buffered in LDS via async-to-LDS, shared by 8 waves.
// ---------------------------------------------------------------------------
__global__ __launch_bounds__(256, 1) void attn_kernel(const _Float16* __restrict__ fT,
                                                      const _Float16* __restrict__ gT,
                                                      const _Float16* __restrict__ hC,
                                                      const float* __restrict__ x,
                                                      const float* __restrict__ gamma,
                                                      float* __restrict__ out) {
    __shared__ _Float16 ldsH[2][256 * 32];    // [c][k]  2 x 16 KB
    __shared__ _Float16 ldsF[2][32 * 32];     // [n][ch] 2 x  2 KB
    __shared__ _Float16 ldsP[8 * 16 * 32];    // per-wave p tile, 8 KB

    const int tid  = threadIdx.x;
    const int lane = tid & 31;
    const int wave = tid >> 5;
    const int gid  = blockIdx.x * 8 + wave;   // 1024 waves = 4 * 256
    const int b    = gid >> 8;
    const int m0   = (gid & 255) << 4;
    const int half = lane >> 4, l16 = lane & 15;

    const _Float16* fTb = fT + (size_t)b * 4096 * 32;
    const _Float16* gTb = gT + (size_t)b * 4096 * 32;
    const _Float16* hb  = hC + (size_t)b * 256 * 4096;

    // A fragment of g^T for this m-tile: [16 m x 32 ch], loaded once
    FragH ag;
#pragma unroll
    for (int j = 0; j < 8; ++j) {
        int k0 = 2 * j + (j >= 4 ? 8 : 0) + (half ? 8 : 0);
        ag.u[j] = *(const unsigned*)(gTb + (size_t)(m0 + l16) * 32 + k0);
    }

    v8f acc[16];
#pragma unroll
    for (int t = 0; t < 16; ++t) acc[t] = (v8f){};
    float mrow[8], lrow[8];
#pragma unroll
    for (int v = 0; v < 8; ++v) { mrow[v] = -3.0e38f; lrow[v] = 0.f; }

    _Float16* myP = ldsP + wave * 16 * 32;

    // prime the pipeline: buffer 0 <- slab n0=0
    stage_async(fTb, hb, &ldsF[0][0], &ldsH[0][0], tid);

    for (int n0 = 0; n0 < 4096; n0 += 32) {
        const int buf = (n0 >> 5) & 1;
        // my async writes to `buf` done, then block-wide: everyone's done
        asm volatile("s_wait_asynccnt 0x0" ::: "memory");
        __syncthreads();
        // overlap: start staging next slab into the other buffer
        if (n0 + 32 < 4096)
            stage_async(fTb + (size_t)(n0 + 32) * 32, hb + (n0 + 32),
                        &ldsF[buf ^ 1][0], &ldsH[buf ^ 1][0], tid);

        const _Float16* Fb = &ldsF[buf][0];
        const _Float16* Hb = &ldsH[buf][0];

        // s = g^T f : two 16x16 tiles (n0, n0+16)
        FragH bf0, bf1;
#pragma unroll
        for (int j = 0; j < 8; ++j) {
            int k0 = 2 * j + (j >= 4 ? 8 : 0) + (half ? 8 : 0);
            bf0.u[j] = *(const unsigned*)(Fb + l16 * 32 + k0);
            bf1.u[j] = *(const unsigned*)(Fb + (16 + l16) * 32 + k0);
        }
        v8f zf = {};
        v8f s0 = __builtin_amdgcn_wmma_f32_16x16x32_f16(false, ag.v, false, bf0.v,
                                                        (short)0, zf, false, false);
        v8f s1 = __builtin_amdgcn_wmma_f32_16x16x32_f16(false, ag.v, false, bf1.v,
                                                        (short)0, zf, false, false);

        // online softmax over this 32-wide slab; rows live in 16-lane halves
#pragma unroll
        for (int v = 0; v < 8; ++v) {
            float tm = fmaxf(s0[v], s1[v]);
            tm = fmaxf(tm, __shfl_xor(tm, 1, 32));
            tm = fmaxf(tm, __shfl_xor(tm, 2, 32));
            tm = fmaxf(tm, __shfl_xor(tm, 4, 32));
            tm = fmaxf(tm, __shfl_xor(tm, 8, 32));
            float mn = fmaxf(mrow[v], tm);
            float sc = __expf(mrow[v] - mn);
            mrow[v] = mn;
            float p0 = __expf(s0[v] - mn);
            float p1 = __expf(s1[v] - mn);
            float ps = p0 + p1;
            ps += __shfl_xor(ps, 1, 32);
            ps += __shfl_xor(ps, 2, 32);
            ps += __shfl_xor(ps, 4, 32);
            ps += __shfl_xor(ps, 8, 32);
            lrow[v] = lrow[v] * sc + ps;
#pragma unroll
            for (int t = 0; t < 16; ++t) acc[t][v] *= sc;
            // write p tile (C/D layout -> LDS row-major 16x32)
            int M = v + 8 * half;
            myP[M * 32 + l16]      = (_Float16)p0;
            myP[M * 32 + 16 + l16] = (_Float16)p1;
        }
        asm volatile("s_wait_dscnt 0x0" ::: "memory");

        // A fragment of p [16 m x 32 n]
        FragH ap;
#pragma unroll
        for (int j = 0; j < 8; ++j) {
            int k0 = 2 * j + (j >= 4 ? 8 : 0) + (half ? 8 : 0);
            ap.u[j] = *(const unsigned*)(myP + l16 * 32 + k0);
        }
        // o += p * h^T over 16 channel tiles
#pragma unroll
        for (int t = 0; t < 16; ++t) {
            FragH bh;
#pragma unroll
            for (int j = 0; j < 8; ++j) {
                int k0 = 2 * j + (j >= 4 ? 8 : 0) + (half ? 8 : 0);
                bh.u[j] = *(const unsigned*)(Hb + (t * 16 + l16) * 32 + k0);
            }
            acc[t] = __builtin_amdgcn_wmma_f32_16x16x32_f16(false, ap.v, false, bh.v,
                                                            (short)0, acc[t], false, false);
        }
    }

    // epilogue: out = gamma * o / l + x
    float gm = gamma[0];
    const float* xb = x + (size_t)b * 256 * 4096;
    float* ob = out + (size_t)b * 256 * 4096;
#pragma unroll
    for (int t = 0; t < 16; ++t) {
#pragma unroll
        for (int v = 0; v < 8; ++v) {
            int m = m0 + v + 8 * half;
            int c = t * 16 + l16;
            float o = acc[t][v] / lrow[v];
            ob[(size_t)c * 4096 + m] = gm * o + xb[(size_t)c * 4096 + m];
        }
    }
}

// ---------------------------------------------------------------------------
extern "C" void kernel_launch(void* const* d_in, const int* in_sizes, int n_in,
                              void* d_out, int out_size, void* d_ws, size_t ws_size,
                              hipStream_t stream) {
    const float* x     = (const float*)d_in[0];
    const float* wf    = (const float*)d_in[1];
    const float* bf    = (const float*)d_in[2];
    const float* wg    = (const float*)d_in[3];
    const float* bg    = (const float*)d_in[4];
    const float* wh    = (const float*)d_in[5];
    const float* bh    = (const float*)d_in[6];
    const float* gamma = (const float*)d_in[7];
    float* out = (float*)d_out;

    char* ws = (char*)d_ws;
    _Float16* Wall = (_Float16*)(ws + OFF_WALL);
    float*    ball = (float*)(ws + OFF_BALL);
    _Float16* xHT  = (_Float16*)(ws + OFF_XHT);
    _Float16* fT   = (_Float16*)(ws + OFF_FT);
    _Float16* gT   = (_Float16*)(ws + OFF_GT);
    _Float16* hC   = (_Float16*)(ws + OFF_HC);

    pack_w_kernel<<<320, 256, 0, stream>>>(wf, bf, wg, bg, wh, bh, Wall, ball);
    xpose_kernel<<<4096, 256, 0, stream>>>(x, xHT);
    proj_kernel<<<2560, 256, 0, stream>>>(Wall, ball, xHT, fT, gT, hC);
    attn_kernel<<<128, 256, 0, stream>>>(fT, gT, hC, x, gamma, out);
}
